// VectorQuantizer_7593502179399
// MI455X (gfx1250) — matrix operand order repaired and checked by reading, not compiled, not run on hardware
//
#include <hip/hip_runtime.h>

// ---------------------------------------------------------------------------
// VQ forward for MI455X (gfx1250, wave32, WMMA):
//   scores = -2 z.E^T + ||e||^2, argmin over 8192 codes per row, fused.
//   Split-bf16 (hi+lo) GEMM on v_wmma_f32_16x16x32_bf16 (3 WMMAs per K=32).
//   B operand staged cooperatively in LDS (double-buffered, async-to-LDS)
//   so each WG reads B from L2 once: 8 GB total vs 32 GB streamed.
// ---------------------------------------------------------------------------

typedef __attribute__((ext_vector_type(16))) __bf16 v16bf;
typedef __attribute__((ext_vector_type(8)))  float  v8f;
typedef int v4i __attribute__((vector_size(16)));   // builtin's pointee type

#define N_ROWS    32768
#define DDIM      512
#define KCODES    8192
#define CHUNKS    16          // DDIM / 32
#define CODETILES 512         // KCODES / 16
#define WAVES_WG  4
#define NCT       2           // code tiles per pipelined group
#define GROUPS    (CODETILES / NCT)        // 256
#define GRP_BYTES (NCT * 2 * CHUNKS * 1024)  // 64 KB per group (hi+lo)
#define CPT       (GRP_BYTES / 16 / 128)     // 32 b128 copies per thread

// workspace byte offsets
#define A_HI_OFF   0ull                 // 2048*16*512 bf16 = 32 MB
#define A_LO_OFF   33554432ull
#define B_HI_OFF   67108864ull          // 8 MB
#define B_LO_OFF   75497472ull          // 8 MB (== B_HI + 8388608)
#define E2_OFF     83886080ull
#define IDXW_OFF   83918848ull

// output layout: [z_q_st 16777216][vq_loss 1][indices 32768]
#define ZQ_ELEMS   16777216
#define LOSS_POS   16777216
#define OUTIDX_POS 16777217

#define AS1 __attribute__((address_space(1)))
#define AS3 __attribute__((address_space(3)))

#if __has_builtin(__builtin_amdgcn_global_load_async_to_lds_b128)
#define USE_ASYNC 1
#else
#define USE_ASYNC 0
#endif

__device__ __forceinline__ void waitAsyncN() {
#if USE_ASYNC
#if __has_builtin(__builtin_amdgcn_s_wait_asynccnt)
    __builtin_amdgcn_s_wait_asynccnt(CPT);
#else
    asm volatile("s_wait_asynccnt 0x20" ::: "memory");
#endif
#endif
}
__device__ __forceinline__ void waitAsync0() {
#if USE_ASYNC
#if __has_builtin(__builtin_amdgcn_s_wait_asynccnt)
    __builtin_amdgcn_s_wait_asynccnt(0);
#else
    asm volatile("s_wait_asynccnt 0x0" ::: "memory");
#endif
#endif
}

// ---------------------------------------------------------------------------
// Pack z into A-fragment layout (ISA 7.12.2, 16-bit A 16x32), split hi/lo bf16.
// lane<16 : K = {0..7,16..23}; lane>=16 : K = {8..15,24..31}
// ---------------------------------------------------------------------------
__global__ void vq_pack_a(const float* __restrict__ z,
                          __bf16* __restrict__ aHi, __bf16* __restrict__ aLo) {
    unsigned gid  = blockIdx.x * blockDim.x + threadIdx.x;   // 1,048,576
    unsigned frag = gid >> 5, lane = gid & 31;
    unsigned rt = frag >> 4, c = frag & 15;
    unsigned m  = rt * 16 + (lane & 15);
    unsigned kb = c * 32 + ((lane >> 4) << 3);
    const float* zr = z + (size_t)m * DDIM;
    size_t fo = (size_t)frag * 512 + lane * 16;
#pragma unroll
    for (int j = 0; j < 16; ++j) {
        unsigned k = kb + j + ((j >> 3) << 3);
        float x = zr[k];
        __bf16 h = (__bf16)x;
        __bf16 l = (__bf16)(x - (float)h);
        aHi[fo + j] = h;
        aLo[fo + j] = l;
    }
}

// ---------------------------------------------------------------------------
// Pack E^T into B-fragment layout (16-bit B 32x16): per lane d-indices of one
// code; lane<16 : K = {0..7,16..23}; lane>=16 : K = {8..15,24..31}. Split hi/lo bf16.
// ---------------------------------------------------------------------------
__global__ void vq_pack_b(const float* __restrict__ E,
                          __bf16* __restrict__ bHi, __bf16* __restrict__ bLo) {
    unsigned gid  = blockIdx.x * blockDim.x + threadIdx.x;   // 262,144
    unsigned frag = gid >> 5, lane = gid & 31;
    unsigned ct = frag >> 4, c = frag & 15;
    unsigned n  = ct * 16 + (lane & 15);
    unsigned koff = c * 32 + ((lane >> 4) << 3);
    const float* er = E + (size_t)n * DDIM + koff;
    size_t fo = (size_t)frag * 512 + lane * 16;
#pragma unroll
    for (int j = 0; j < 16; ++j) {
        float x = er[j + ((j >> 3) << 3)];
        __bf16 h = (__bf16)x;
        __bf16 l = (__bf16)(x - (float)h);
        bHi[fo + j] = h;
        bLo[fo + j] = l;
    }
}

// ---------------------------------------------------------------------------
// ||e_k||^2 per code, plus zero the loss accumulator in d_out.
// ---------------------------------------------------------------------------
__global__ void vq_e2(const float* __restrict__ E, float* __restrict__ e2,
                      float* __restrict__ lossSlot) {
    unsigned k = blockIdx.x * blockDim.x + threadIdx.x;      // 8192
    if (k == 0) *lossSlot = 0.0f;
    if (k < KCODES) {
        const float4* r = (const float4*)(E + (size_t)k * DDIM);
        float s = 0.0f;
        for (int i = 0; i < DDIM / 4; ++i) {
            float4 v = r[i];
            s += v.x * v.x + v.y * v.y + v.z * v.z + v.w * v.w;
        }
        e2[k] = s;
    }
}

// ---------------------------------------------------------------------------
// Fused GEMM + argmin. 4 waves/WG; wave w owns row tile blockIdx*4+w, all
// 8192 codes. A frags staged once in LDS (lane-private). B frags staged
// cooperatively per group of 2 code tiles, double-buffered (async-to-LDS).
// acc init -0.5*||e||^2 => argmin(dist) == argmax(acc). Even/odd-chunk
// accumulator split gives 4 independent WMMA chains for the XDL pipe.
// ---------------------------------------------------------------------------
__global__ __launch_bounds__(128)
void vq_gemm_argmin(const __bf16* __restrict__ aHi, const __bf16* __restrict__ aLo,
                    const char* __restrict__ bBase,   // bHi; bLo at +8388608
                    const float* __restrict__ e2, int* __restrict__ outIdx) {
    __shared__ __bf16 ldsA[WAVES_WG * 2 * CHUNKS * 512];          // 128 KB
    __shared__ __align__(16) __bf16 ldsB[2 * NCT * 2 * CHUNKS * 512]; // 128 KB
    __shared__ float  redMax[WAVES_WG * 32 * 8];                  // 4 KB
    __shared__ int    redIdx[WAVES_WG * 32 * 8];                  // 4 KB

    const int tid  = threadIdx.x;
    const int wave = tid >> 5, lane = tid & 31;
    const int rt   = blockIdx.x * WAVES_WG + wave;
    const int nlane = lane & 15;

    // ---- stage this wave's A fragments (hi+lo) into LDS, lane-private ----
    {
        const size_t gb = ((size_t)rt * CHUNKS) * 512 + lane * 16;
        __bf16* dh = &ldsA[(size_t)((wave * 2 + 0) * CHUNKS) * 512 + lane * 16];
        __bf16* dl = &ldsA[(size_t)((wave * 2 + 1) * CHUNKS) * 512 + lane * 16];
#pragma unroll
        for (int c = 0; c < CHUNKS; ++c) {
            *(v16bf*)(dh + c * 512) = *(const v16bf*)(aHi + gb + (size_t)c * 512);
            *(v16bf*)(dl + c * 512) = *(const v16bf*)(aLo + gb + (size_t)c * 512);
        }
    }

    // ---- cooperative B-group copy: 64 KB (2 code tiles, hi+lo, 16 chunks) --
    auto issueGroup = [&](int grp, int buf) {
        const size_t ct0 = (size_t)grp * NCT;
        char* lb = (char*)ldsB + (size_t)buf * GRP_BYTES;
#pragma unroll
        for (int i = 0; i < CPT; ++i) {
            unsigned o = (unsigned)(tid + i * 128) * 16u;   // byte in group
            unsigned r = o >> 14;                           // region: u*2+hl
            unsigned hl = r & 1, u = r >> 1;
            unsigned inner = o & 16383u;
            const char* src = bBase + (size_t)hl * 8388608ull
                                    + (ct0 + u) * 16384ull + inner;
            char* dst = lb + (size_t)r * 16384ull + inner;
#if USE_ASYNC
            __builtin_amdgcn_global_load_async_to_lds_b128(
                (AS1 v4i*)(uintptr_t)src,
                (AS3 v4i*)(void*)dst, 0, 0);
#else
            *(uint4*)dst = *(const uint4*)src;
#endif
        }
    };

    issueGroup(0, 0);   // prologue fill of buffer 0

    const __bf16* lHi = &ldsA[(size_t)((wave * 2 + 0) * CHUNKS) * 512 + lane * 16];
    const __bf16* lLo = &ldsA[(size_t)((wave * 2 + 1) * CHUNKS) * 512 + lane * 16];

    float runMax[8];
    int   runIdx[8];
#pragma unroll
    for (int g = 0; g < 8; ++g) { runMax[g] = -3.4e38f; runIdx[g] = 0; }

    for (int grp = 0; grp < GROUPS; ++grp) {
        const int buf = grp & 1;
        if (grp + 1 < GROUPS) { issueGroup(grp + 1, buf ^ 1); waitAsyncN(); }
        else                  { waitAsync0(); }
        __syncthreads();    // group `grp` resident in ldsB[buf]

        v8f acc[NCT][2];
#pragma unroll
        for (int u = 0; u < NCT; ++u) {
            float c0 = -0.5f * e2[(grp * NCT + u) * 16 + nlane];
#pragma unroll
            for (int g = 0; g < 8; ++g) { acc[u][0][g] = c0; acc[u][1][g] = 0.0f; }
        }

        const __bf16* lbB = ldsB + (size_t)buf * (NCT * 2 * CHUNKS * 512) + lane * 16;
#pragma unroll 2
        for (int c = 0; c < CHUNKS; ++c) {
            v16bf Ah = *(const v16bf*)(lHi + c * 512);
            v16bf Al = *(const v16bf*)(lLo + c * 512);
            const int p = c & 1;
#pragma unroll
            for (int u = 0; u < NCT; ++u) {
                const __bf16* bb = lbB + (size_t)((u * 2 + 0) * CHUNKS + c) * 512;
                v16bf Bh = *(const v16bf*)bb;
                v16bf Bl = *(const v16bf*)(bb + (size_t)CHUNKS * 512);
                acc[u][p] = __builtin_amdgcn_wmma_f32_16x16x32_bf16(
                                false, Ah, false, Bh, (short)0, acc[u][p], false, false);
                acc[u][p] = __builtin_amdgcn_wmma_f32_16x16x32_bf16(
                                false, Ah, false, Bl, (short)0, acc[u][p], false, false);
                acc[u][p] = __builtin_amdgcn_wmma_f32_16x16x32_bf16(
                                false, Al, false, Bh, (short)0, acc[u][p], false, false);
            }
        }

        // running argmax; codes ascend per lane => first-wins tie handling
#pragma unroll
        for (int u = 0; u < NCT; ++u) {
            int code = (grp * NCT + u) * 16 + nlane;
#pragma unroll
            for (int g = 0; g < 8; ++g) {
                float v = acc[u][0][g] + acc[u][1][g];
                if (v > runMax[g]) { runMax[g] = v; runIdx[g] = code; }
            }
        }
        __syncthreads();    // everyone done with ldsB[buf] before refill
    }

    // ---- cross-lane reduction: row m = g + 8*(lane>=16), 16 lanes/row ----
#pragma unroll
    for (int g = 0; g < 8; ++g) {
        redMax[(wave * 32 + lane) * 8 + g] = runMax[g];
        redIdx[(wave * 32 + lane) * 8 + g] = runIdx[g];
    }
    __syncthreads();
    if (lane < 16) {
        int r = lane, h = r >> 3, g = r & 7;
        float bm = -3.4e38f; int bi = 0x7fffffff;
        for (int l2 = 0; l2 < 16; ++l2) {
            int src = wave * 32 + h * 16 + l2;
            float v = redMax[src * 8 + g];
            int   i = redIdx[src * 8 + g];
            if (v > bm || (v == bm && i < bi)) { bm = v; bi = i; }
        }
        outIdx[rt * 16 + r] = bi;
    }
}

// ---------------------------------------------------------------------------
// Epilogue: z_q gather (straight-through forward value == z_q), index output,
// vq_loss = 1.25 * mean((z_q - z_e)^2) via block reduce + float atomic.
// ---------------------------------------------------------------------------
__global__ void vq_epilogue(const float* __restrict__ z, const float* __restrict__ E,
                            const int* __restrict__ idxw, float* __restrict__ out) {
    __shared__ float red[8];
    unsigned gid = blockIdx.x * blockDim.x + threadIdx.x;    // 131072
    if (gid < N_ROWS) out[OUTIDX_POS + gid] = (float)idxw[gid];

    float acc = 0.0f;
    const size_t stride = (size_t)gridDim.x * blockDim.x;
    for (size_t e = gid; e < (size_t)ZQ_ELEMS; e += stride) {
        unsigned row = (unsigned)(e >> 9);
        unsigned d   = (unsigned)(e & 511);
        int k = idxw[row];
        float q  = E[(size_t)k * DDIM + d];
        float ze = z[e];
        out[e] = q;
        float df = q - ze;
        acc += df * df;
    }
    for (int off = 16; off > 0; off >>= 1) acc += __shfl_xor(acc, off, 32);
    int lane = threadIdx.x & 31, w = threadIdx.x >> 5;
    if (lane == 0) red[w] = acc;
    __syncthreads();
    if (threadIdx.x == 0) {
        float t = 0.0f;
        for (int i = 0; i < 8; ++i) t += red[i];
        atomicAdd(out + LOSS_POS, t * (1.25f / 16777216.0f));
    }
}

// ---------------------------------------------------------------------------
extern "C" void kernel_launch(void* const* d_in, const int* in_sizes, int n_in,
                              void* d_out, int out_size, void* d_ws, size_t ws_size,
                              hipStream_t stream) {
    const float* z = (const float*)d_in[0];     // [8,4096,512]
    const float* E = (const float*)d_in[1];     // [8192,512]
    float* out = (float*)d_out;
    char*  ws  = (char*)d_ws;

    __bf16* aHi = (__bf16*)(ws + A_HI_OFF);
    __bf16* aLo = (__bf16*)(ws + A_LO_OFF);
    __bf16* bHi = (__bf16*)(ws + B_HI_OFF);
    __bf16* bLo = (__bf16*)(ws + B_LO_OFF);
    float*  e2  = (float*) (ws + E2_OFF);
    int*    idw = (int*)   (ws + IDXW_OFF);

    vq_pack_b<<<1024, 256, 0, stream>>>(E, bHi, bLo);
    vq_pack_a<<<4096, 256, 0, stream>>>(z, aHi, aLo);
    vq_e2<<<32, 256, 0, stream>>>(E, e2, out + LOSS_POS);
    vq_gemm_argmin<<<512, 128, 0, stream>>>(aHi, aLo, (const char*)(ws + B_HI_OFF),
                                            e2, idw);
    vq_epilogue<<<512, 256, 0, stream>>>(z, E, idw, out);
}
